// ConvMHSA_78168404787413
// MI455X (gfx1250) — compile-verified, matrix operand order checked by reading, not run on hardware
//
#include <hip/hip_runtime.h>

typedef __attribute__((ext_vector_type(16))) _Float16 v16h;
typedef __attribute__((ext_vector_type(8)))  float    v8f;
typedef unsigned int u32;
typedef u32 u32x4 __attribute__((ext_vector_type(4)));
typedef int i32x4 __attribute__((ext_vector_type(4)));
typedef int i32x8 __attribute__((ext_vector_type(8)));

#if __has_include(<hip/amd_detail/amd_gfx1250_TDM.h>)
#define TDM_6ARG 1
#endif

#define C_DIM 384
#define NH    6
#define HD    64
#define HIN   56
#define WIN   56
#define NQ    (HIN * WIN)     /* 3136 */
#define NKV   (28 * 28)       /* 784  */
#define BATCH 8
#define SPAD  800             /* 784 padded to multiple of 32 */

static __device__ __forceinline__ int lane_id() { return threadIdx.x & 31; }

// K-offset inside a 32-wide slab for the 16-bit A/B fragment layout:
// pairs 0..3 -> k = 2p + 8*half ; pairs 4..7 -> k = 16 + 2(p-4) + 8*half
static __device__ __forceinline__ int frag_k(int p, int half) {
    int base = (p < 4) ? (p * 2) : (16 + (p - 4) * 2);
    return base + half * 8;
}

// A fragment (16x32, f16) from row-major f32, rows = lane&15, ld in floats.
// Also used for B = W^T fragments (W is (N,K) row-major: B(k,n) = W[n*ld + k]).
static __device__ __forceinline__ v16h load_frag_rowmajor(const float* __restrict__ src, int ld) {
    int l = lane_id();
    int m = l & 15, half = l >> 4;
    const float* row = src + m * ld;
    v16h a;
#pragma unroll
    for (int p = 0; p < 8; ++p) {
        int k = frag_k(p, half);
        float2 f = *(const float2*)(row + k);
        a[2 * p]     = (_Float16)f.x;
        a[2 * p + 1] = (_Float16)f.y;
    }
    return a;
}

// Fragment from an LDS-resident row-major f32 slab with leading dim 32.
static __device__ __forceinline__ v16h load_frag_lds_row32(const float* S) {
    int l = lane_id();
    int m = l & 15, half = l >> 4;
    const float* row = S + m * 32;
    v16h a;
#pragma unroll
    for (int p = 0; p < 8; ++p) {
        int k = frag_k(p, half);
        a[2 * p]     = (_Float16)row[k];
        a[2 * p + 1] = (_Float16)row[k + 1];
    }
    return a;
}

// A fragment from LDS f32 tile (row-major, ld floats), K slab at k0.
static __device__ __forceinline__ v16h load_frag_lds(const float* S, int ld, int k0) {
    int l = lane_id();
    int m = l & 15, half = l >> 4;
    v16h a;
#pragma unroll
    for (int p = 0; p < 8; ++p) {
        int k = k0 + frag_k(p, half);
        a[2 * p]     = (_Float16)S[m * ld + k];
        a[2 * p + 1] = (_Float16)S[m * ld + k + 1];
    }
    return a;
}

// B fragment (32x16, f16) where B(k,n) = src[(k0+k)*ld + n], rows >= kmax read as 0.
static __device__ __forceinline__ v16h load_frag_colmajor_guard(const float* __restrict__ src,
                                                                int ld, int k0, int kmax) {
    int l = lane_id();
    int n = l & 15, half = l >> 4;
    v16h b;
#pragma unroll
    for (int p = 0; p < 8; ++p) {
        int k = k0 + frag_k(p, half);
        b[2 * p]     = (k < kmax)     ? (_Float16)src[k * ld + n]       : (_Float16)0.f;
        b[2 * p + 1] = (k + 1 < kmax) ? (_Float16)src[(k + 1) * ld + n] : (_Float16)0.f;
    }
    return b;
}

static __device__ __forceinline__ v8f wmma16(v16h a, v16h b, v8f c) {
    return __builtin_amdgcn_wmma_f32_16x16x32_f16(false, a, false, b, (short)0, c, false, false);
}

// ---- Tensor Data Mover: 2D tile (tile_n rows x tile_k cols of f32, row stride
// ---- ld elements) from global into LDS at lds_addr (byte LDS address).
// ---- D# layout per CDNA5 ISA 8.3/8.4: group0 = {flags, lds_addr, gaddr, type=2},
// ---- group1 = {data_size=4B, tensor dims, tile dims, dim0 stride}.
static __device__ __forceinline__ void tdm_load_2d(u32 lds_addr, const float* gsrc,
                                                   int tile_k, int tile_n, int ld) {
    unsigned long long ga = (unsigned long long)(size_t)gsrc;
    u32x4 g0;
    g0.x = 1u;                                            // count=1, user descriptor
    g0.y = lds_addr;                                      // LDS byte address
    g0.z = (u32)(ga & 0xffffffffu);                       // global_addr[31:0]
    g0.w = (u32)((ga >> 32) & 0x1ffffffu) | (2u << 30);   // global_addr[56:32], type=2
    i32x8 g1;
    g1[0] = (2 << 16);                                    // data_size = 2 -> 4 bytes
    g1[1] = (ld & 0xffff) << 16;                          // tensor_dim0[15:0]
    g1[2] = ((ld >> 16) & 0xffff) | ((tile_n & 0xffff) << 16);  // dim0 hi | tensor_dim1 lo
    g1[3] = (tile_k & 0xffff) << 16;                      // tensor_dim1 hi=0 | tile_dim0
    g1[4] = (tile_n & 0xffff);                            // tile_dim1 | tile_dim2=0
    g1[5] = ld;                                           // tensor_dim0_stride[31:0]
    g1[6] = 0;                                            // stride hi | dim1_stride lo
    g1[7] = 0;
    i32x4 z4 = {0, 0, 0, 0};
#ifdef TDM_6ARG
    i32x8 z8 = {0, 0, 0, 0, 0, 0, 0, 0};
    __builtin_amdgcn_tensor_load_to_lds(g0, g1, z4, z4, z8, 0);
#else
    __builtin_amdgcn_tensor_load_to_lds(g0, g1, z4, z4, 0);
#endif
}

// ---- Fused QKV GEMM: q/k/v = x @ W^T + b. Block = 8 waves covering 128 rows at a
// ---- fixed 16-col weight panel; B slabs TDM-staged into LDS, double-buffered.
__global__ __launch_bounds__(256) void qkv_gemm_tdm_kernel(
        const float* __restrict__ X,
        const float* __restrict__ Wq, const float* __restrict__ bq,
        const float* __restrict__ Wk, const float* __restrict__ bk,
        const float* __restrict__ Wv, const float* __restrict__ bv,
        float* __restrict__ Oq, float* __restrict__ Ok, float* __restrict__ Ov,
        int M) {
    __shared__ float Bs[2][3][16 * 32];   // 12 KB, double-buffered B slabs for q/k/v
    const int K = C_DIM, N = C_DIM;
    const int ntiles = N / 16;            // 24
    const int nt = blockIdx.x % ntiles;
    const int mg = blockIdx.x / ntiles;
    const int wave = threadIdx.x >> 5;
    const int mt = mg * 8 + wave;

    const float* Ab  = X  + (long)(mt * 16) * K;
    const float* Wqb = Wq + (long)(nt * 16) * K;
    const float* Wkb = Wk + (long)(nt * 16) * K;
    const float* Wvb = Wv + (long)(nt * 16) * K;

    if (threadIdx.x < 32) {               // wave 0 drives the TDM (EXEC-independent op)
        tdm_load_2d((u32)(size_t)&Bs[0][0][0], Wqb, 32, 16, K);
        tdm_load_2d((u32)(size_t)&Bs[0][1][0], Wkb, 32, 16, K);
        tdm_load_2d((u32)(size_t)&Bs[0][2][0], Wvb, 32, 16, K);
    }

    v8f cq = {}, ck = {}, cv = {};
    const int steps = K / 32;             // 12
    for (int it = 0; it < steps; ++it) {
        const int k0 = it * 32;
        if (threadIdx.x < 32) {
            if (it + 1 < steps) {         // prefetch next slab, then wait for current
                const int nb = (it + 1) & 1;
                tdm_load_2d((u32)(size_t)&Bs[nb][0][0], Wqb + k0 + 32, 32, 16, K);
                tdm_load_2d((u32)(size_t)&Bs[nb][1][0], Wkb + k0 + 32, 32, 16, K);
                tdm_load_2d((u32)(size_t)&Bs[nb][2][0], Wvb + k0 + 32, 32, 16, K);
                __builtin_amdgcn_s_wait_tensorcnt(3);
            } else {
                __builtin_amdgcn_s_wait_tensorcnt(0);
            }
        }
        __syncthreads();                  // current slab visible to all waves
        const int cb = it & 1;
        __builtin_prefetch(Ab + k0 + 128, 0, 1);   // global_prefetch_b8 for A stream
        v16h a   = load_frag_rowmajor(Ab + k0, K);
        v16h bqf = load_frag_lds_row32(&Bs[cb][0][0]);
        v16h bkf = load_frag_lds_row32(&Bs[cb][1][0]);
        v16h bvf = load_frag_lds_row32(&Bs[cb][2][0]);
        cq = wmma16(a, bqf, cq);
        ck = wmma16(a, bkf, ck);
        cv = wmma16(a, bvf, cv);
        __syncthreads();                  // all reads done before buffer is re-filled
    }

    int l = lane_id(), n = l & 15, half = l >> 4;
#pragma unroll
    for (int r = 0; r < 8; ++r) {
        long m  = (long)(mt * 16 + r + half * 8);
        int  nn = nt * 16 + n;
        Oq[m * N + nn] = cq[r] + bq[nn];
        Ok[m * N + nn] = ck[r] + bk[nn];
        Ov[m * N + nn] = cv[r] + bv[nn];
    }
}

// ------------------ GEMM: out[m,n] = sum_k A[m,k] * W[n,k] + bias[n] -------------
__global__ __launch_bounds__(256) void gemm_bias_kernel(const float* __restrict__ A,
                                                        const float* __restrict__ W,
                                                        const float* __restrict__ bias,
                                                        float* __restrict__ out,
                                                        int M, int N, int K) {
    int wave   = blockIdx.x * (blockDim.x >> 5) + (threadIdx.x >> 5);
    int ntiles = N >> 4;
    int tiles  = (M >> 4) * ntiles;
    if (wave >= tiles) return;                 // wave-uniform: EXEC stays all-1s
    int mt = wave / ntiles, nt = wave % ntiles;

    const float* Ab = A + (long)(mt * 16) * K;
    const float* Bb = W + (long)(nt * 16) * K;

    v8f c = {};
    for (int k0 = 0; k0 < K; k0 += 32) {
        __builtin_prefetch(Ab + k0 + 128, 0, 1);   // global_prefetch_b8
        __builtin_prefetch(Bb + k0 + 128, 0, 1);
        v16h a = load_frag_rowmajor(Ab + k0, K);
        v16h b = load_frag_rowmajor(Bb + k0, K);
        c = wmma16(a, b, c);
    }
    int l = lane_id(), n = l & 15, half = l >> 4;
#pragma unroll
    for (int r = 0; r < 8; ++r) {
        int m  = mt * 16 + r + half * 8;
        int nn = nt * 16 + n;
        out[(long)m * N + nn] = c[r] + bias[nn];
    }
}

// ------------------ depthwise 3x3 conv on channels-last tokens -------------------
__global__ __launch_bounds__(256) void dwconv_kernel(const float* __restrict__ in,
                                                     const float* __restrict__ w,
                                                     const float* __restrict__ bias,
                                                     float* __restrict__ out,
                                                     int stride) {
    int Ho = (HIN + 2 - 3) / stride + 1;
    int Wo = (WIN + 2 - 3) / stride + 1;
    long total = (long)BATCH * Ho * Wo * C_DIM;
    long t = (long)blockIdx.x * blockDim.x + threadIdx.x;
    if (t >= total) return;
    int c  = (int)(t % C_DIM);
    long r = t / C_DIM;
    int ox = (int)(r % Wo); r /= Wo;
    int oy = (int)(r % Ho);
    int b  = (int)(r / Ho);

    float acc = bias[c];
#pragma unroll
    for (int ky = 0; ky < 3; ++ky) {
#pragma unroll
        for (int kx = 0; kx < 3; ++kx) {
            int iy = oy * stride - 1 + ky;
            int ix = ox * stride - 1 + kx;
            if (iy >= 0 && iy < HIN && ix >= 0 && ix < WIN) {
                acc += in[((long)b * NQ + iy * WIN + ix) * C_DIM + c] * w[c * 9 + ky * 3 + kx];
            }
        }
    }
    out[t] = acc;
}

// ------------------ fused attention: S = scale*Q K^T, neg_softmax, O = P V ------
__global__ __launch_bounds__(256) void attn_kernel(const float* __restrict__ Q,
                                                   const float* __restrict__ Kf,
                                                   const float* __restrict__ V,
                                                   float* __restrict__ O) {
    __shared__ float S[16 * SPAD];     // 51.2 KB score/prob tile
    __shared__ float red[16 * 16];
    __shared__ float rowmax[16];
    __shared__ float rowsum[16];

    const int qt = blockIdx.x;         // q tile (16 rows), 196 of them
    const int h  = blockIdx.y;         // head
    const int b  = blockIdx.z;         // batch
    const int tid  = threadIdx.x;
    const int wave = tid >> 5;
    const int l = tid & 31, n = l & 15, half = l >> 4;
    const float scale = 0.125f;        // hd^-0.5, hd=64

    // ---- Phase 1: S tile = scale * q kT, 49 k-tiles spread over 8 waves ----
    const float* qbase = Q + ((long)(b * NQ + qt * 16)) * C_DIM + h * HD;
    v16h a0 = load_frag_rowmajor(qbase, C_DIM);
    v16h a1 = load_frag_rowmajor(qbase + 32, C_DIM);
    for (int kt = wave; kt < NKV / 16; kt += 8) {
        const float* kbase = Kf + ((long)(b * NKV + kt * 16)) * C_DIM + h * HD;
        v16h b0 = load_frag_rowmajor(kbase, C_DIM);       // B(d,n) = K[n,d]
        v16h b1 = load_frag_rowmajor(kbase + 32, C_DIM);
        v8f c = {};
        c = wmma16(a0, b0, c);
        c = wmma16(a1, b1, c);
#pragma unroll
        for (int r = 0; r < 8; ++r)
            S[(r + half * 8) * SPAD + kt * 16 + n] = c[r] * scale;
    }
    // zero-pad columns 784..799 (sign(0)=0 => padded probs are exactly 0)
    S[(tid >> 4) * SPAD + NKV + (tid & 15)] = 0.f;
    __syncthreads();

    // ---- Phase 2: neg_softmax per row (16 slices per row) ----
    const int row = tid >> 4, slice = tid & 15;
    float mx = 0.f;
    for (int col = slice; col < NKV; col += 16)
        mx = fmaxf(mx, fabsf(S[row * SPAD + col]));
    red[row * 16 + slice] = mx;
    __syncthreads();
    if (tid < 16) {
        float m = 0.f;
#pragma unroll
        for (int i = 0; i < 16; ++i) m = fmaxf(m, red[tid * 16 + i]);
        rowmax[tid] = m;
    }
    __syncthreads();
    float rm = rowmax[row];
    float sum = 0.f;
    for (int col = slice; col < NKV; col += 16) {
        float s  = S[row * SPAD + col];
        float sg = (s > 0.f) ? 1.f : ((s < 0.f) ? -1.f : 0.f);
        float e  = __expf(fabsf(s) - rm);
        S[row * SPAD + col] = sg * e;
        sum += e;
    }
    red[row * 16 + slice] = sum;
    __syncthreads();
    if (tid < 16) {
        float s = 0.f;
#pragma unroll
        for (int i = 0; i < 16; ++i) s += red[tid * 16 + i];
        rowsum[tid] = s;
    }
    __syncthreads();
    float inv = 1.f / rowsum[row];
    for (int col = slice; col < NKV; col += 16)
        S[row * SPAD + col] *= inv;
    __syncthreads();

    // ---- Phase 3: O(16x64) = P(16x800) @ V(800x64); waves 0..3 own 16 cols each ----
    if (wave < 4) {                    // wave-uniform branch
        const float* vbase = V + (long)(b * NKV) * C_DIM + h * HD + wave * 16;
        v8f c = {};
        for (int k0 = 0; k0 < SPAD; k0 += 32) {
            v16h pa = load_frag_lds(S, SPAD, k0);
            v16h vb = load_frag_colmajor_guard(vbase, C_DIM, k0, NKV);
            c = wmma16(pa, vb, c);
        }
#pragma unroll
        for (int r = 0; r < 8; ++r) {
            long m = (long)(b * NQ + qt * 16 + r + half * 8);
            O[m * C_DIM + h * HD + wave * 16 + n] = c[r];
        }
    }
}

// --------------------------------- launcher -------------------------------------
extern "C" void kernel_launch(void* const* d_in, const int* in_sizes, int n_in,
                              void* d_out, int out_size, void* d_ws, size_t ws_size,
                              hipStream_t stream) {
    const float* x      = (const float*)d_in[0];
    const float* q_w    = (const float*)d_in[3];
    const float* q_b    = (const float*)d_in[4];
    const float* k_w    = (const float*)d_in[5];
    const float* k_b    = (const float*)d_in[6];
    const float* v_w    = (const float*)d_in[7];
    const float* v_b    = (const float*)d_in[8];
    const float* q_dw_w = (const float*)d_in[9];
    const float* q_dw_b = (const float*)d_in[10];
    const float* k_dw_w = (const float*)d_in[11];
    const float* k_dw_b = (const float*)d_in[12];
    const float* v_dw_w = (const float*)d_in[13];
    const float* v_dw_b = (const float*)d_in[14];
    const float* proj_w = (const float*)d_in[15];
    const float* proj_b = (const float*)d_in[16];
    float* out = (float*)d_out;

    const long sizeQ  = (long)BATCH * NQ  * C_DIM;   // 9,633,792 floats
    const long sizeKV = (long)BATCH * NKV * C_DIM;   // 2,408,448 floats
    float* ws = (float*)d_ws;
    float* q_lin = ws;                       // B,NQ,C
    float* k_lin = ws + sizeQ;               // B,NQ,C  (reused later for attn out)
    float* v_lin = ws + 2 * sizeQ;           // B,NQ,C
    float* q_cv  = ws + 3 * sizeQ;           // B,NQ,C
    float* k_cv  = ws + 4 * sizeQ;           // B,NKV,C
    float* v_cv  = k_cv + sizeKV;            // B,NKV,C
    float* a_out = k_lin;                    // reuse k_lin buffer

    const int M = BATCH * NQ;                // 25088 (multiple of 16)

    // 1) Fused QKV projection: one pass over x, 3 WMMA accumulators per wave,
    //    TDM-staged weight panels in LDS.
    {
        const int blocks = (M / 16 / 8) * (C_DIM / 16);   // 196 * 24 = 4704
        qkv_gemm_tdm_kernel<<<blocks, 256, 0, stream>>>(
            x, q_w, q_b, k_w, k_b, v_w, v_b, q_lin, k_lin, v_lin, M);
    }

    // 2) depthwise 3x3 convs (q stride 1; k,v stride 2)
    {
        long tq = (long)BATCH * NQ * C_DIM;
        long tk = (long)BATCH * NKV * C_DIM;
        dwconv_kernel<<<(int)((tq + 255) / 256), 256, 0, stream>>>(q_lin, q_dw_w, q_dw_b, q_cv, 1);
        dwconv_kernel<<<(int)((tk + 255) / 256), 256, 0, stream>>>(k_lin, k_dw_w, k_dw_b, k_cv, 2);
        dwconv_kernel<<<(int)((tk + 255) / 256), 256, 0, stream>>>(v_lin, v_dw_w, v_dw_b, v_cv, 2);
    }

    // 3) fused attention with neg_softmax (score tile lives in LDS)
    {
        dim3 grid(NQ / 16, NH, BATCH);       // (196, 6, 8)
        attn_kernel<<<grid, 256, 0, stream>>>(q_cv, k_cv, v_cv, a_out);
    }

    // 4) output projection
    {
        const int tiles  = (M / 16) * (C_DIM / 16);
        const int blocks = (tiles + 7) / 8;
        gemm_bias_kernel<<<blocks, 256, 0, stream>>>(a_out, proj_w, proj_b, out, M, C_DIM, C_DIM);
    }
}
